// Net_29326036697839
// MI455X (gfx1250) — matrix operand order repaired and checked by reading, not compiled, not run on hardware
//
#include <hip/hip_runtime.h>
#include <hip/hip_bf16.h>
#include <math.h>

typedef __attribute__((ext_vector_type(16))) _Float16 v16h;
typedef __attribute__((ext_vector_type(8)))  _Float16 v8h;
typedef __attribute__((ext_vector_type(8)))  float    v8f;

#define KSP   5
#define K3    125
#define NNODE 10000
#define NEDGE 160000
#define NCLS  6890
#define NCLSP 6912   // NCLS padded to multiple of 64

// ---------------- utility kernels ----------------

__global__ void zero_f32(float* p, size_t n) {
  size_t i = (size_t)blockIdx.x * blockDim.x + threadIdx.x;
  if (i < n) p[i] = 0.f;
}

// dense B [Kd, Nc] -> panel layout: panel p (16 cols), stored [16][Kd] col-major,
// padded with zeros up to Ncpad columns.
__global__ void conv_dense_panel_k(const float* __restrict__ src, _Float16* __restrict__ dst,
                                   int Kd, int Nc, int Ncpad) {
  size_t i = (size_t)blockIdx.x * blockDim.x + threadIdx.x;
  size_t tot = (size_t)Kd * Ncpad;
  if (i >= tot) return;
  int panel = (int)(i / ((size_t)16 * Kd));
  int rem   = (int)(i % ((size_t)16 * Kd));
  int c  = rem / Kd;
  int kk = rem % Kd;
  int j  = panel * 16 + c;
  dst[i] = (j < Nc) ? (_Float16)src[(size_t)kk * Nc + j] : (_Float16)0.0f;
}

// spline W [K3, In, Out] -> panel layout over columns j = k*Out + o, Kd = In
__global__ void conv_splineW_panel_k(const float* __restrict__ W, _Float16* __restrict__ dst,
                                     int In, int Out) {
  size_t i = (size_t)blockIdx.x * blockDim.x + threadIdx.x;
  size_t tot = (size_t)K3 * Out * In;
  if (i >= tot) return;
  int panel = (int)(i / ((size_t)16 * In));
  int rem   = (int)(i % ((size_t)16 * In));
  int c  = rem / In;
  int ii = rem % In;
  int j  = panel * 16 + c;      // global column, < K3*Out (multiple of 16)
  int k3 = j / Out;
  int oo = j % Out;
  dst[i] = (_Float16)W[((size_t)k3 * In + ii) * Out + oo];
}

// ---------------- spline basis ----------------

__global__ void basis_k(const float* pseudo, float* bw, int* flat, int E) {
  int e = blockIdx.x * blockDim.x + threadIdx.x;
  if (e >= E) return;
  float w0[3], w1[3]; int lo[3], hi[3];
  #pragma unroll
  for (int d = 0; d < 3; ++d) {
    float p  = pseudo[e * 3 + d] * (float)(KSP - 1);
    float lf = floorf(p);
    float fr = p - lf;
    int l = (int)lf; l = l < 0 ? 0 : (l > KSP - 1 ? KSP - 1 : l);
    int h = l + 1;   h = h > KSP - 1 ? KSP - 1 : h;
    w0[d] = 1.f - fr; w1[d] = fr; lo[d] = l; hi[d] = h;
  }
  #pragma unroll
  for (int s = 0; s < 8; ++s) {
    int ci = (s >> 2) & 1, cj = (s >> 1) & 1, ck = s & 1;
    float bb = (ci ? w1[0] : w0[0]) * (cj ? w1[1] : w0[1]) * (ck ? w1[2] : w0[2]);
    int fl = (ci ? hi[0] : lo[0]) + KSP * (cj ? hi[1] : lo[1]) + KSP * KSP * (ck ? hi[2] : lo[2]);
    bw[e * 8 + s] = bb;
    flat[e * 8 + s] = fl;
  }
}

__global__ void degree_k(const int* dstv, float* deg, int E) {
  int e = blockIdx.x * blockDim.x + threadIdx.x;
  if (e < E) atomicAdd(&deg[dstv[e]], 1.f);
}

// ---------------- WMMA GEMM:  C[M,Nc] = A[M,Kd] * B[Kd,Nc]  (f16 in, f32 acc) ----------------
// Bf is in 16-column panel layout [npanels][16][Kd] (zero-padded columns).
// EPI 0: store f16 raw | EPI 1: f16, +bias, ELU | EPI 2: f32, +bias
template <int EPI>
__global__ void wmma_gemm(const _Float16* __restrict__ A, const _Float16* __restrict__ Bf,
                          void* __restrict__ Cv, const float* __restrict__ bias,
                          int M, int Nc, int Kd, int lda, int ldc) {
  const int lane = threadIdx.x & 31;
  const int wave = threadIdx.x >> 5;
  const int m0 = blockIdx.x * 16;
  const int n0 = (blockIdx.y * 4 + wave) * 16;
  if (n0 >= Nc || m0 >= M) return;

  const int r16 = lane & 15;   // A-row within tile / B,C column within tile
  const int hID = lane >> 4;   // lane half

  const _Float16* Ap = A + (size_t)(m0 + r16) * lda;              // lane's A row
  const _Float16* Bp = Bf + (size_t)(n0 >> 4) * 16 * Kd + (size_t)r16 * Kd;  // lane's B column

  v8f acc = {};
  for (int kb = 0; kb < Kd; kb += 32) {
    if (kb + 32 < Kd) {
      __builtin_prefetch(Ap + kb + 32, 0, 0);
      __builtin_prefetch(Bp + kb + 32, 0, 0);
    }
    // A 16x32 f16 fragment: lanes0-15 K=0-7/16-23, lanes16-31 K=8-15/24-31 (ISA 7.12.2)
    v8h alo = *(const v8h*)(Ap + kb + hID * 8);
    v8h ahi = *(const v8h*)(Ap + kb + 16 + hID * 8);
    v16h a = __builtin_shufflevector(alo, ahi, 0, 1, 2, 3, 4, 5, 6, 7,
                                     8, 9, 10, 11, 12, 13, 14, 15);
    // B 32x16 f16 fragment: lanes0-15 K=0-7/16-23, lanes16-31 K=8-15/24-31 (matches A)
    v8h blo = *(const v8h*)(Bp + kb + hID * 8);
    v8h bhi = *(const v8h*)(Bp + kb + 16 + hID * 8);
    v16h b = __builtin_shufflevector(blo, bhi, 0, 1, 2, 3, 4, 5, 6, 7,
                                     8, 9, 10, 11, 12, 13, 14, 15);
    acc = __builtin_amdgcn_wmma_f32_16x16x32_f16(false, a, false, b, (short)0, acc, false, false);
  }

  const int col = n0 + r16;
  if (col < Nc) {
    float bv = (EPI > 0) ? bias[col] : 0.f;
    #pragma unroll
    for (int r = 0; r < 8; ++r) {
      int rr = m0 + hID * 8 + r;        // C: VGPR r -> row r / 8+r (ISA 7.12.2)
      float v = acc[r];
      if (EPI > 0)  v += bv;
      if (EPI == 1) v = v > 0.f ? v : (expf(v) - 1.f);
      if (EPI == 2) ((float*)Cv)[(size_t)rr * ldc + col] = v;
      else          ((_Float16*)Cv)[(size_t)rr * ldc + col] = (_Float16)v;
    }
  }
}

// ---------------- layer-1 special case (In == 1): xW[n,k,o] = x[n]*W[k,o] ----------------

__global__ void xw_in1_k(const float* x, const float* W, _Float16* xw,
                         int klo, int ksz, int Out, int N) {
  size_t i = (size_t)blockIdx.x * blockDim.x + threadIdx.x;
  size_t tot = (size_t)N * ksz * Out;
  if (i >= tot) return;
  int n   = (int)(i / ((size_t)ksz * Out));
  int rem = (int)(i % ((size_t)ksz * Out));
  xw[i] = (_Float16)(x[n] * W[(size_t)klo * Out + rem]);
}

// ---------------- edge gather + scatter ----------------

__global__ void edge_msg_k(const _Float16* __restrict__ xw, const int* __restrict__ srcv,
                           const int* __restrict__ dstv, const float* __restrict__ bw,
                           const int* __restrict__ flat, float* __restrict__ agg,
                           int E, int Out, int klo, int khi) {
  size_t t = (size_t)blockIdx.x * blockDim.x + threadIdx.x;
  int e = (int)(t / Out);
  int o = (int)(t % Out);
  if (e >= E) return;
  int s0 = srcv[e], d0 = dstv[e];
  int ksz = khi - klo;
  float m = 0.f;
  bool any = false;
  #pragma unroll
  for (int s = 0; s < 8; ++s) {
    int k = flat[e * 8 + s];
    if (k >= klo && k < khi) {
      m += bw[e * 8 + s] * (float)xw[((size_t)s0 * ksz + (k - klo)) * Out + o];
      any = true;
    }
  }
  if (any) atomicAdd(&agg[(size_t)d0 * Out + o], m);
}

// ---------------- finalize: agg/deg + h@root + bias, ELU ----------------

__global__ void finalize_k(const float* __restrict__ agg, const float* __restrict__ deg,
                           const float* __restrict__ hp32, const float* __restrict__ root,
                           const float* __restrict__ bias, float* ho32, _Float16* ho16,
                           int In, int Out) {
  __shared__ float sh[64];
  int n = blockIdx.x, o = threadIdx.x;
  if (o < In) sh[o] = hp32[(size_t)n * In + o];
  __syncthreads();
  float r = 0.f;
  for (int i = 0; i < In; ++i) r += sh[i] * root[(size_t)i * Out + o];
  float d = deg[n]; d = d > 1.f ? d : 1.f;
  float v = agg[(size_t)n * Out + o] / d + r + bias[o];
  v = v > 0.f ? v : (expf(v) - 1.f);
  ho32[(size_t)n * Out + o] = v;
  ho16[(size_t)n * Out + o] = (_Float16)v;
}

// ---------------- in-place log-softmax over rows of NCLS ----------------

__global__ void logsoftmax_k(float* out, int Ncls) {
  __shared__ float sm[256];
  int n = blockIdx.x;
  float* row = out + (size_t)n * Ncls;
  float mx = -INFINITY;
  for (int c = threadIdx.x; c < Ncls; c += blockDim.x) mx = fmaxf(mx, row[c]);
  sm[threadIdx.x] = mx; __syncthreads();
  for (int s = blockDim.x / 2; s > 0; s >>= 1) {
    if ((int)threadIdx.x < s) sm[threadIdx.x] = fmaxf(sm[threadIdx.x], sm[threadIdx.x + s]);
    __syncthreads();
  }
  mx = sm[0]; __syncthreads();
  float sum = 0.f;
  for (int c = threadIdx.x; c < Ncls; c += blockDim.x) sum += expf(row[c] - mx);
  sm[threadIdx.x] = sum; __syncthreads();
  for (int s = blockDim.x / 2; s > 0; s >>= 1) {
    if ((int)threadIdx.x < s) sm[threadIdx.x] += sm[threadIdx.x + s];
    __syncthreads();
  }
  float lse = mx + logf(sm[0]);
  for (int c = threadIdx.x; c < Ncls; c += blockDim.x) row[c] = row[c] - lse;
}

// ---------------- host orchestration ----------------

extern "C" void kernel_launch(void* const* d_in, const int* in_sizes, int n_in,
                              void* d_out, int out_size, void* d_ws, size_t ws_size,
                              hipStream_t stream) {
  (void)in_sizes; (void)n_in; (void)out_size;
  const float* x      = (const float*)d_in[0];
  const int*   ei     = (const int*)d_in[1];
  const float* pseudo = (const float*)d_in[2];
  const float* W1     = (const float*)d_in[3];
  const float* root1  = (const float*)d_in[4];
  const float* b1     = (const float*)d_in[5];
  const float* W2     = (const float*)d_in[6];
  const float* root2  = (const float*)d_in[7];
  const float* b2     = (const float*)d_in[8];
  const float* Wsb    = (const float*)d_in[9];
  const float* rootsb = (const float*)d_in[10];
  const float* bsb    = (const float*)d_in[11];
  const float* l1w    = (const float*)d_in[12];
  const float* l1b    = (const float*)d_in[13];
  const float* l2w    = (const float*)d_in[14];
  const float* l2b    = (const float*)d_in[15];
  const int* src = ei;
  const int* dst = ei + NEDGE;

  char* ws = (char*)d_ws;
  size_t off = 0;
  auto alloc = [&](size_t bytes) -> char* {
    char* p = ws + off;
    off += (bytes + 255) & ~(size_t)255;
    return p;
  };
  float*    bw   = (float*)alloc((size_t)NEDGE * 8 * 4);
  int*      flat = (int*)  alloc((size_t)NEDGE * 8 * 4);
  float*    deg  = (float*)alloc((size_t)NNODE * 4);
  float*    agg  = (float*)alloc((size_t)NNODE * 64 * 4);
  float*    hA32 = (float*)alloc((size_t)NNODE * 64 * 4);
  float*    hB32 = (float*)alloc((size_t)NNODE * 64 * 4);
  _Float16* hA16 = (_Float16*)alloc((size_t)NNODE * 64 * 2);
  _Float16* hB16 = (_Float16*)alloc((size_t)NNODE * 64 * 2);
  _Float16* h256 = (_Float16*)alloc((size_t)NNODE * 256 * 2);
  _Float16* w16  = (_Float16*)alloc((size_t)256 * NCLSP * 2);
  _Float16* xw   = (_Float16*)(ws + off);
  size_t xwcap_halves = (ws_size > off) ? (ws_size - off) / 2 : (size_t)NNODE * 64;

  auto run_gemm = [&](int epi, const _Float16* A, const _Float16* B, void* C, const float* bias,
                      int M, int Nc, int Kd, int lda, int ldc) {
    dim3 g((M + 15) / 16, (Nc + 63) / 64);
    dim3 blk(128);
    if (epi == 0)      wmma_gemm<0><<<g, blk, 0, stream>>>(A, B, C, bias, M, Nc, Kd, lda, ldc);
    else if (epi == 1) wmma_gemm<1><<<g, blk, 0, stream>>>(A, B, C, bias, M, Nc, Kd, lda, ldc);
    else               wmma_gemm<2><<<g, blk, 0, stream>>>(A, B, C, bias, M, Nc, Kd, lda, ldc);
  };

  // basis + degree (shared across all spline layers)
  basis_k<<<(NEDGE + 255) / 256, 256, 0, stream>>>(pseudo, bw, flat, NEDGE);
  zero_f32<<<(NNODE + 255) / 256, 256, 0, stream>>>(deg, (size_t)NNODE);
  degree_k<<<(NEDGE + 255) / 256, 256, 0, stream>>>(dst, deg, NEDGE);

  auto spline = [&](const float* hp32, const _Float16* hp16, int In, int Out,
                    const float* W, const float* root, const float* bias,
                    float* ho32, _Float16* ho16) {
    size_t naggr = (size_t)NNODE * Out;
    zero_f32<<<(unsigned)((naggr + 255) / 256), 256, 0, stream>>>(agg, naggr);
    if (In > 1) {
      size_t nW = (size_t)K3 * Out * In;
      conv_splineW_panel_k<<<(unsigned)((nW + 255) / 256), 256, 0, stream>>>(W, w16, In, Out);
    }
    size_t perk = (size_t)NNODE * Out;  // halves per k-slot of xW
    int kc = K3;
    if (xwcap_halves < (size_t)K3 * perk) {
      kc = (int)(xwcap_halves / perk);
      if (kc < 1) kc = 1;
    }
    for (int klo = 0; klo < K3; klo += kc) {
      int ksz = (K3 - klo) < kc ? (K3 - klo) : kc;
      if (In == 1) {
        size_t tot = (size_t)NNODE * ksz * Out;
        xw_in1_k<<<(unsigned)((tot + 255) / 256), 256, 0, stream>>>(hp32, W, xw, klo, ksz, Out, NNODE);
      } else {
        // panel offset for columns starting at klo*Out (multiple of 16): klo*Out*In halves
        run_gemm(0, hp16, w16 + (size_t)klo * Out * In, xw, nullptr,
                 NNODE, ksz * Out, In, In, ksz * Out);
      }
      size_t tot = (size_t)NEDGE * Out;
      edge_msg_k<<<(unsigned)((tot + 255) / 256), 256, 0, stream>>>(
          xw, src, dst, bw, flat, agg, NEDGE, Out, klo, klo + ksz);
    }
    finalize_k<<<NNODE, Out, 0, stream>>>(agg, deg, hp32, root, bias, ho32, ho16, In, Out);
  };

  // 6 spline-conv layers (ping-pong h buffers)
  spline(x,    nullptr, 1, 32, W1, root1, b1, hA32, hA16);
  spline(hA32, hA16,   32, 64, W2, root2, b2, hB32, hB16);
  const size_t WSZ = (size_t)K3 * 64 * 64, RSZ = 64 * 64;
  spline(hB32, hB16, 64, 64, Wsb + 0 * WSZ, rootsb + 0 * RSZ, bsb + 0 * 64, hA32, hA16);
  spline(hA32, hA16, 64, 64, Wsb + 1 * WSZ, rootsb + 1 * RSZ, bsb + 1 * 64, hB32, hB16);
  spline(hB32, hB16, 64, 64, Wsb + 2 * WSZ, rootsb + 2 * RSZ, bsb + 2 * 64, hA32, hA16);
  spline(hA32, hA16, 64, 64, Wsb + 3 * WSZ, rootsb + 3 * RSZ, bsb + 3 * 64, hB32, hB16);

  // lin1: [N,64]x[64,256] + bias, ELU -> f16
  conv_dense_panel_k<<<(unsigned)(((size_t)64 * 256 + 255) / 256), 256, 0, stream>>>(
      l1w, w16, 64, 256, 256);
  run_gemm(1, hB16, w16, h256, l1b, NNODE, 256, 64, 64, 256);

  // lin2: [N,256]x[256,6890] + bias -> f32 logits in d_out (B padded to 6912 cols)
  conv_dense_panel_k<<<(unsigned)(((size_t)256 * NCLSP + 255) / 256), 256, 0, stream>>>(
      l2w, w16, 256, NCLS, NCLSP);
  run_gemm(2, h256, w16, d_out, l2b, NNODE, NCLS, 256, 256, NCLS);

  // log-softmax in place
  logsoftmax_k<<<NNODE, 256, 0, stream>>>((float*)d_out, NCLS);
}
